// SelfAttention_74225624809935
// MI455X (gfx1250) — matrix operand order repaired and checked by reading, not compiled, not run on hardware
//
#include <hip/hip_runtime.h>
#include <stdint.h>

typedef __attribute__((ext_vector_type(16))) _Float16 v16h;
typedef __attribute__((ext_vector_type(8)))  _Float16 v8h;
typedef __attribute__((ext_vector_type(8)))  float    v8f;
typedef __attribute__((ext_vector_type(4)))  float    v4f;

#define Bv   4
#define Tv   2048
#define Cv   1024
#define TP   2048
#define TT   4096   // T_PAST + T

// ---------------- WMMA helpers (layouts per CDNA5 ISA 7.12.2) ----

__device__ __forceinline__ v8f wmma16(v16h a, v16h b, v8f c) {
  // D = A(16x32 f16) * B(32x16 f16) + C(16x16 f32)
  return __builtin_amdgcn_wmma_f32_16x16x32_f16(false, a, false, b, (short)0, c,
                                                false, false);
}

__device__ __forceinline__ v16h comb16(v8h lo, v8h hi) {
  v16h t;
#pragma unroll
  for (int j = 0; j < 8; ++j) { t[j] = lo[j]; t[8 + j] = hi[j]; }
  return t;
}

// A fragment, f16 row-major source.  Lanes 0-15 hold K {0..7,16..23},
// lanes 16-31 hold K {8..15,24..31}.  p = row_start + k + (laneHi?8:0).
__device__ __forceinline__ v16h load_a_f16(const _Float16* p) {
  v8h h0 = *(const v8h*)(p);       // K lo+0..7
  v8h h1 = *(const v8h*)(p + 16);  // K lo+16..23
  return comb16(h0, h1);
}

// B fragment, f16 source with K contiguous for this lane's column.
// Lanes 0-15 hold K {0..7,16..23}, lanes 16-31 hold K {8..15,24..31}; N = lane%16.
// p = col_row_start + k + (laneHi?8:0)
__device__ __forceinline__ v16h load_b_f16(const _Float16* p) {
  v8h h0 = *(const v8h*)(p);
  v8h h1 = *(const v8h*)(p + 16);
  return comb16(h0, h1);
}

__device__ __forceinline__ void wait_asynccnt0() {
#if __has_builtin(__builtin_amdgcn_s_wait_asynccnt)
  __builtin_amdgcn_s_wait_asynccnt(0);
#else
  asm volatile("s_wait_asynccnt 0x0" ::: "memory");
#endif
}

// ---------------- Kernel 0: bulk f32 -> f16 convert ----------------------
__global__ __launch_bounds__(256) void cvt_f32_f16(const float* __restrict__ src,
                                                   _Float16* __restrict__ dst) {
  const long long i = ((long long)blockIdx.x * 256 + threadIdx.x) * 8;
  v4f f0 = *(const v4f*)(src + i);
  v4f f1 = *(const v4f*)(src + i + 4);
  v8h h;
#pragma unroll
  for (int j = 0; j < 4; ++j) {
    h[j]     = (_Float16)f0[j];
    h[4 + j] = (_Float16)f1[j];
  }
  *(v8h*)(dst + i) = h;
}

// ---------------- Kernel 1: QKV projection (all-f16 operands) ------------
// y[row, n] = sum_c x[row, c] * W[n, c] + bias[n]   (row = b*T + t, 8192 rows)
// Wave tile 64x32 (4 M-tiles x 2 N-tiles), 8 waves (2x4) -> 128x128 per block.
// MODE 0: write f16 row-major (Q -> ws).  MODE 1: write f32 into concat K/V.
template <int MODE>
__global__ __launch_bounds__(256) void qkv_gemm(
    const _Float16* __restrict__ xh, const _Float16* __restrict__ wh,
    const float* __restrict__ bias, float* __restrict__ dst_f32,
    _Float16* __restrict__ dst_f16) {
  const int lane = threadIdx.x & 31;
  const int wave = threadIdx.x >> 5;
  const int l16  = lane & 15;
  const int hi   = lane >> 4;          // 0 | 1
  const int rowBase = blockIdx.x * 128 + (wave >> 2) * 64;
  const int colBase = blockIdx.y * 128 + (wave & 3) * 32;

  v8f zero = {0.f, 0.f, 0.f, 0.f, 0.f, 0.f, 0.f, 0.f};
  v8f acc[4][2];
#pragma unroll
  for (int mt = 0; mt < 4; ++mt)
#pragma unroll
    for (int nt = 0; nt < 2; ++nt) acc[mt][nt] = zero;

  for (int k = 0; k < Cv; k += 32) {
    v16h a[4];
#pragma unroll
    for (int mt = 0; mt < 4; ++mt)
      a[mt] = load_a_f16(xh + (size_t)(rowBase + mt * 16 + l16) * Cv + k + hi * 8);
    v16h bf[2];
#pragma unroll
    for (int nt = 0; nt < 2; ++nt)
      bf[nt] = load_b_f16(wh + (size_t)(colBase + nt * 16 + l16) * Cv + k + hi * 8);
#pragma unroll
    for (int mt = 0; mt < 4; ++mt)
#pragma unroll
      for (int nt = 0; nt < 2; ++nt) acc[mt][nt] = wmma16(a[mt], bf[nt], acc[mt][nt]);
  }

  const float bv0 = bias[colBase + l16];
  const float bv1 = bias[colBase + 16 + l16];
#pragma unroll
  for (int mt = 0; mt < 4; ++mt) {
#pragma unroll
    for (int r = 0; r < 8; ++r) {
      const int row = rowBase + mt * 16 + hi * 8 + r;  // global row = b*T + t
      if (MODE == 0) {
        _Float16* dp = dst_f16 + (size_t)row * Cv + colBase + l16;
        dp[0]  = (_Float16)(acc[mt][0][r] + bv0);
        dp[16] = (_Float16)(acc[mt][1][r] + bv1);
      } else {
        const int b = row >> 11, t = row & (Tv - 1);   // K/V -> concat out
        float* dp = dst_f32 + ((size_t)b * TT + TP + t) * Cv + colBase + l16;
        dp[0]  = acc[mt][0][r] + bv0;
        dp[16] = acc[mt][1][r] + bv1;
      }
    }
  }
}

// ---------------- Kernel 2: scores = Q @ past_k^T (f16 x f16) ------------
__global__ __launch_bounds__(256) void scores_gemm(
    const _Float16* __restrict__ q, const _Float16* __restrict__ kh,
    float* __restrict__ scores) {
  const int i0 = blockIdx.x * 128;
  const int s0 = blockIdx.y * 128;
  if (s0 > i0 + 127) return;  // fully above the causal diagonal: skip
  const int b = blockIdx.z;
  const int lane = threadIdx.x & 31;
  const int wave = threadIdx.x >> 5;
  const int l16 = lane & 15;
  const int hi  = lane >> 4;
  const int rowBase = i0 + (wave >> 2) * 64;
  const int colBase = s0 + (wave & 3) * 32;
  const _Float16* qb = q + (size_t)b * Tv * Cv;
  const _Float16* kb = kh + (size_t)b * TP * Cv;

  v8f zero = {0.f, 0.f, 0.f, 0.f, 0.f, 0.f, 0.f, 0.f};
  v8f acc[4][2];
#pragma unroll
  for (int mt = 0; mt < 4; ++mt)
#pragma unroll
    for (int nt = 0; nt < 2; ++nt) acc[mt][nt] = zero;

  for (int k = 0; k < Cv; k += 32) {
    v16h a[4];
#pragma unroll
    for (int mt = 0; mt < 4; ++mt)
      a[mt] = load_a_f16(qb + (size_t)(rowBase + mt * 16 + l16) * Cv + k + hi * 8);
    v16h bf[2];
#pragma unroll
    for (int nt = 0; nt < 2; ++nt)
      bf[nt] = load_b_f16(kb + (size_t)(colBase + nt * 16 + l16) * Cv + k + hi * 8);
#pragma unroll
    for (int mt = 0; mt < 4; ++mt)
#pragma unroll
      for (int nt = 0; nt < 2; ++nt) acc[mt][nt] = wmma16(a[mt], bf[nt], acc[mt][nt]);
  }

#pragma unroll
  for (int mt = 0; mt < 4; ++mt)
#pragma unroll
    for (int r = 0; r < 8; ++r) {
      const int row = rowBase + mt * 16 + hi * 8 + r;
      float* dp = scores + ((size_t)b * Tv + row) * TP + colBase + l16;
      dp[0]  = acc[mt][0][r];
      dp[16] = acc[mt][1][r];
    }
}

// ---------------- Kernel 3: masked softmax -> f16 weights ----------------
__global__ __launch_bounds__(256) void softmax_kernel(
    const float* __restrict__ scores, _Float16* __restrict__ wts) {
  const int row = blockIdx.x;                 // 0..8191
  const int b = row >> 11, i = row & (Tv - 1);
  const float* srow = scores + ((size_t)b * Tv + i) * TP;
  _Float16* wrow = wts + ((size_t)b * Tv + i) * TP;
  const int nvalid = i + 1;                   // mask: j <= i (past keys only)
  const float scale = 0.03125f;               // 1/sqrt(1024)

  __shared__ float red[256];
  float m = -3.0e38f;
  for (int s = threadIdx.x; s < nvalid; s += 256) m = fmaxf(m, srow[s] * scale);
  red[threadIdx.x] = m;
  __syncthreads();
  for (int off = 128; off; off >>= 1) {
    if ((int)threadIdx.x < off)
      red[threadIdx.x] = fmaxf(red[threadIdx.x], red[threadIdx.x + off]);
    __syncthreads();
  }
  m = red[0];
  __syncthreads();

  float sum = 0.f;
  for (int s = threadIdx.x; s < nvalid; s += 256) sum += __expf(srow[s] * scale - m);
  red[threadIdx.x] = sum;
  __syncthreads();
  for (int off = 128; off; off >>= 1) {
    if ((int)threadIdx.x < off) red[threadIdx.x] += red[threadIdx.x + off];
    __syncthreads();
  }
  const float inv = 1.0f / red[0];

  for (int s = threadIdx.x; s < TP; s += 256) {
    const float w = (s < nvalid) ? __expf(srow[s] * scale - m) * inv : 0.0f;
    wrow[s] = (_Float16)w;  // zeros above diagonal keep WMMA tiles exact
  }
}

// ---------------- Kernel 3.5: past_v -> f16 V^T --------------------------
// Stages the 32x32 f32 tile into LDS with CDNA5 async global->LDS DMA
// (GLOBAL_LOAD_ASYNC_TO_LDS_B32, ASYNCcnt-tracked), then reads it back
// transposed and emits f16.
__global__ __launch_bounds__(256) void transpose_v(const float* __restrict__ pv,
                                                   _Float16* __restrict__ vt) {
  __shared__ float tile[32][33];
  const int b = blockIdx.z;
  const int s0 = blockIdx.x * 32, c0 = blockIdx.y * 32;
  const float* src = pv + (size_t)b * TP * Cv;
  _Float16* dst = vt + (size_t)b * Cv * TP;

#pragma unroll
  for (int rr = 0; rr < 4; ++rr) {
    const int r = rr * 8 + threadIdx.y;
    const float* g = src + (size_t)(s0 + r) * Cv + c0 + threadIdx.x;
    const uint32_t ldsoff = (uint32_t)(uintptr_t)(&tile[r][threadIdx.x]);
    asm volatile("global_load_async_to_lds_b32 %0, %1, off"
                 :: "v"(ldsoff), "v"(g) : "memory");
  }
  wait_asynccnt0();
  __syncthreads();

#pragma unroll
  for (int rr = 0; rr < 4; ++rr) {
    const int r = rr * 8 + threadIdx.y;
    dst[(size_t)(c0 + r) * TP + s0 + threadIdx.x] = (_Float16)tile[threadIdx.x][r];
  }
}

// ---------------- Kernel 4: out = weights @ past_v -----------------------
__global__ __launch_bounds__(256) void out_gemm(const _Float16* __restrict__ wts,
                                                const _Float16* __restrict__ vt,
                                                float* __restrict__ out) {
  const int b = blockIdx.z;
  const int i0 = blockIdx.x * 128;
  const int lane = threadIdx.x & 31;
  const int wave = threadIdx.x >> 5;
  const int l16 = lane & 15;
  const int hi  = lane >> 4;
  const int rowBase = i0 + (wave >> 2) * 64;
  const int colBase = blockIdx.y * 128 + (wave & 3) * 32;
  const int kend = (i0 + 128 < TP) ? (i0 + 128) : TP;  // triangular: s <= i
  const _Float16* wb = wts + (size_t)b * Tv * TP;
  const _Float16* vb = vt + (size_t)b * Cv * TP;

  v8f zero = {0.f, 0.f, 0.f, 0.f, 0.f, 0.f, 0.f, 0.f};
  v8f acc[4][2];
#pragma unroll
  for (int mt = 0; mt < 4; ++mt)
#pragma unroll
    for (int nt = 0; nt < 2; ++nt) acc[mt][nt] = zero;

  for (int k = 0; k < kend; k += 32) {
    v16h a[4];
#pragma unroll
    for (int mt = 0; mt < 4; ++mt)
      a[mt] = load_a_f16(wb + (size_t)(rowBase + mt * 16 + l16) * TP + k + hi * 8);
    v16h bf[2];
#pragma unroll
    for (int nt = 0; nt < 2; ++nt)
      bf[nt] = load_b_f16(vb + (size_t)(colBase + nt * 16 + l16) * TP + k + hi * 8);
#pragma unroll
    for (int mt = 0; mt < 4; ++mt)
#pragma unroll
      for (int nt = 0; nt < 2; ++nt) acc[mt][nt] = wmma16(a[mt], bf[nt], acc[mt][nt]);
  }

#pragma unroll
  for (int mt = 0; mt < 4; ++mt)
#pragma unroll
    for (int r = 0; r < 8; ++r) {
      const int row = rowBase + mt * 16 + hi * 8 + r;
      float* dp = out + ((size_t)b * Tv + row) * Cv + colBase + l16;
      dp[0]  = acc[mt][0][r];
      dp[16] = acc[mt][1][r];
    }
}

// ---------------- Host launcher ------------------------------------------
extern "C" void kernel_launch(void* const* d_in, const int* in_sizes, int n_in,
                              void* d_out, int out_size, void* d_ws, size_t ws_size,
                              hipStream_t stream) {
  const float* x  = (const float*)d_in[0];
  const float* pk = (const float*)d_in[1];
  const float* pv = (const float*)d_in[2];
  const float* Wq = (const float*)d_in[3];
  const float* bq = (const float*)d_in[4];
  const float* Wk = (const float*)d_in[5];
  const float* bk = (const float*)d_in[6];
  const float* Wv = (const float*)d_in[7];
  const float* bv = (const float*)d_in[8];

  float* out  = (float*)d_out;                       // (B, T, C)
  float* Kout = out + (size_t)Bv * Tv * Cv;          // (B, TT, C)
  float* Vout = Kout + (size_t)Bv * TT * Cv;         // (B, TT, C)

  // 128 MiB workspace with lifetime-based aliasing:
  //  [0,64Mi)   scores f32 (phase 2+)  -- hosts xh(16Mi)+whq/whk/whv(3x2Mi) in phase 1
  //  [64,80Mi)  Q f16 (phase 1-2)      -- reused as V^T f16 (phase 3+)
  //  [80,112Mi) softmax weights f16
  //  [112,128Mi) past_k f16
  char* ws = (char*)d_ws;
  float*    sc   = (float*)(ws);
  _Float16* xh   = (_Float16*)(ws);
  _Float16* whq  = (_Float16*)(ws + ((size_t)16 << 20));
  _Float16* whk  = (_Float16*)(ws + ((size_t)18 << 20));
  _Float16* whv  = (_Float16*)(ws + ((size_t)20 << 20));
  _Float16* qf16 = (_Float16*)(ws + ((size_t)64 << 20));
  _Float16* vt   = (_Float16*)(ws + ((size_t)64 << 20));
  _Float16* wts  = (_Float16*)(ws + ((size_t)80 << 20));
  _Float16* kh   = (_Float16*)(ws + ((size_t)112 << 20));

  // Concatenate: past K/V -> output rows [0, T_PAST)
  for (int b = 0; b < Bv; ++b) {
    hipMemcpyAsync(Kout + (size_t)b * TT * Cv, pk + (size_t)b * TP * Cv,
                   (size_t)TP * Cv * sizeof(float), hipMemcpyDeviceToDevice, stream);
    hipMemcpyAsync(Vout + (size_t)b * TT * Cv, pv + (size_t)b * TP * Cv,
                   (size_t)TP * Cv * sizeof(float), hipMemcpyDeviceToDevice, stream);
  }

  // Phase 0: hoisted f32->f16 conversions (one streaming pass each)
  cvt_f32_f16<<<dim3(Bv * Tv * Cv / 2048), 256, 0, stream>>>(x, xh);
  cvt_f32_f16<<<dim3(Cv * Cv / 2048), 256, 0, stream>>>(Wq, whq);
  cvt_f32_f16<<<dim3(Cv * Cv / 2048), 256, 0, stream>>>(Wk, whk);
  cvt_f32_f16<<<dim3(Cv * Cv / 2048), 256, 0, stream>>>(Wv, whv);
  cvt_f32_f16<<<dim3(Bv * TP * Cv / 2048), 256, 0, stream>>>(pk, kh);

  // Phase 1: QKV projections (pure b128-load + wmma inner loops)
  qkv_gemm<0><<<dim3(Bv * Tv / 128, Cv / 128), 256, 0, stream>>>(xh, whq, bq, nullptr, qf16);
  qkv_gemm<1><<<dim3(Bv * Tv / 128, Cv / 128), 256, 0, stream>>>(xh, whk, bk, Kout, nullptr);
  qkv_gemm<1><<<dim3(Bv * Tv / 128, Cv / 128), 256, 0, stream>>>(xh, whv, bv, Vout, nullptr);

  // Phase 2: triangular scores (overwrites xh/wh region -- both dead now)
  scores_gemm<<<dim3(Tv / 128, TP / 128, Bv), 256, 0, stream>>>(qf16, kh, sc);

  // Phase 3: masked softmax -> f16 weights
  softmax_kernel<<<dim3(Bv * Tv), 256, 0, stream>>>(sc, wts);

  // Phase 3.5: V^T via async global->LDS staging (overwrites dead Q f16)
  transpose_v<<<dim3(TP / 32, Cv / 32, Bv), dim3(32, 8), 0, stream>>>(pv, vt);

  // Phase 4: triangular out = weights @ V
  out_gemm<<<dim3(Tv / 128, Cv / 128, Bv), 256, 0, stream>>>(wts, vt, out);
}
